// SemanticIDQuantizer_42838003811020
// MI455X (gfx1250) — matrix operand order repaired and checked by reading, not compiled, not run on hardware
//
#include <hip/hip_runtime.h>
#include <hip/hip_bf16.h>

// SemanticID RVQ quantizer for MI455X (gfx1250, wave32).
// Memory-bound on 537MB logits writes; all GEMM flops go through
// v_wmma_f32_16x16x32_bf16 with a bf16 hi+lo 3-term split for ~f32 accuracy.

#define BATCH 16384
#define HID   512
#define KCB   2048
#define LQ    4

typedef __attribute__((ext_vector_type(16))) __bf16 v16bf;
typedef __attribute__((ext_vector_type(8)))  __bf16 v8bf;
typedef __attribute__((ext_vector_type(8)))  float  v8f;

// ---- A-operand loader: 16-bit A 16x32 layout ----
// lane l in [0,16): row=l,  K chunks {k0..k0+7} and {k0+16..k0+23}
// lane l in [16,32): row=l-16, K chunks {k0+8..k0+15} and {k0+24..k0+31}
__device__ __forceinline__ v16bf load_a16(const __bf16* __restrict__ row,
                                          int k0, int half) {
  const v8bf c0 = *reinterpret_cast<const v8bf*>(row + k0 + half * 8);
  const v8bf c1 = *reinterpret_cast<const v8bf*>(row + k0 + 16 + half * 8);
  return __builtin_shufflevector(c0, c1, 0, 1, 2, 3, 4, 5, 6, 7,
                                 8, 9, 10, 11, 12, 13, 14, 15);
}

// ---- B-operand loader: 16-bit B 32x16 layout ----
// lane l in [0,16): col=l, K chunks {k0..k0+7} and {k0+16..k0+23}; lanes 16-31: {k0+8..k0+15} and {k0+24..k0+31}
__device__ __forceinline__ v16bf load_b16(const __bf16* __restrict__ colrow,
                                          int k0, int half) {
  const v8bf c0 = *reinterpret_cast<const v8bf*>(colrow + k0 + half * 8);
  const v8bf c1 = *reinterpret_cast<const v8bf*>(colrow + k0 + 16 + half * 8);
  return __builtin_shufflevector(c0, c1, 0, 1, 2, 3, 4, 5, 6, 7,
                                 8, 9, 10, 11, 12, 13, 14, 15);
}

// ---------------- f32 -> bf16 hi/lo split ----------------
__global__ __launch_bounds__(256) void convert_hilo_kernel(
    const float* __restrict__ in, __bf16* __restrict__ hi,
    __bf16* __restrict__ lo, int n) {
  int i = blockIdx.x * 256 + threadIdx.x;
  if (i < n) {
    float x = in[i];
    __bf16 h = (__bf16)x;
    hi[i] = h;
    lo[i] = (__bf16)(x - (float)h);
  }
}

// ---------------- codeword squared norms (exact f32) ----------------
__global__ __launch_bounds__(256) void cnorm_kernel(
    const float* __restrict__ cb, float* __restrict__ cnorm) {
  const int wave = threadIdx.x >> 5, lane = threadIdx.x & 31;
  const int cw = blockIdx.x * 8 + wave;  // L*K codewords total
  const float* p = cb + (long)cw * HID;
  float s = 0.f;
  for (int d = lane; d < HID; d += 32) s += p[d] * p[d];
  #pragma unroll
  for (int off = 16; off > 0; off >>= 1) s += __shfl_xor(s, off, 32);
  if (lane == 0) cnorm[cw] = s;
}

// ---------------- projection: x = f @ W^T + b (WMMA, split bf16) ----
__global__ __launch_bounds__(256) void proj_kernel(
    const __bf16* __restrict__ f_hi, const __bf16* __restrict__ f_lo,
    const __bf16* __restrict__ w_hi, const __bf16* __restrict__ w_lo,
    const float* __restrict__ bias,
    float* __restrict__ residual, __bf16* __restrict__ res_hi,
    __bf16* __restrict__ res_lo) {
  const int tid = threadIdx.x;
  const int wave = tid >> 5, lane = tid & 31;
  const int half = lane >> 4, l15 = lane & 15;
  const int waveM = wave >> 2, waveN = wave & 3;  // 2x4 tiles -> 32x128
  const int rowT = blockIdx.x * 32 + waveM * 16;
  const int colT = blockIdx.y * 128 + waveN * 16;
  const int dcol = colT + l15;

  const __bf16* arh = f_hi + (long)(rowT + l15) * HID;
  const __bf16* arl = f_lo + (long)(rowT + l15) * HID;
  const __bf16* brh = w_hi + (long)dcol * HID;
  const __bf16* brl = w_lo + (long)dcol * HID;

  v8f acc = {};
  #pragma unroll 2
  for (int k0 = 0; k0 < HID; k0 += 32) {
    v16bf ah = load_a16(arh, k0, half);
    v16bf al = load_a16(arl, k0, half);
    v16bf bh = load_b16(brh, k0, half);
    v16bf bl = load_b16(brl, k0, half);
    acc = __builtin_amdgcn_wmma_f32_16x16x32_bf16(false, ah, false, bh,
                                                  (short)0, acc, false, false);
    acc = __builtin_amdgcn_wmma_f32_16x16x32_bf16(false, ah, false, bl,
                                                  (short)0, acc, false, false);
    acc = __builtin_amdgcn_wmma_f32_16x16x32_bf16(false, al, false, bh,
                                                  (short)0, acc, false, false);
  }
  const float bv = bias[dcol];
  #pragma unroll
  for (int j = 0; j < 8; ++j) {
    const int r = rowT + j + 8 * half;  // C layout: VGPR j -> M=j / M=j+8
    const float x = acc[j] + bv;
    const long g = (long)r * HID + dcol;
    residual[g] = x;
    __bf16 h = (__bf16)x;
    res_hi[g] = h;
    res_lo[g] = (__bf16)(x - (float)h);
  }
}

// ---------------- one RVQ level: dists -> logits, argmin, update ----
__global__ __launch_bounds__(256) void rvq_level_kernel(
    const float* __restrict__ codebooks,  // f32 [L][K][HID]
    const __bf16* __restrict__ cb_hi, const __bf16* __restrict__ cb_lo,
    const float* __restrict__ cnorm,  // [L][K]
    float* __restrict__ residual, __bf16* __restrict__ res_hi,
    __bf16* __restrict__ res_lo,
    float* __restrict__ logits,  // [B][L][K]
    float* __restrict__ qsum,    // [B][HID]
    int level) {
  __shared__ unsigned long long s_key[32];
  __shared__ float s_rn[32];
  const int tid = threadIdx.x;
  const int wave = tid >> 5, lane = tid & 31;
  const int half = lane >> 4, l15 = lane & 15;
  const int rowBlk = blockIdx.x * 32;

  if (tid < 32) s_key[tid] = ~0ull;
  __syncthreads();

  // exact f32 row norms, deterministic shuffle-tree (8 lanes per row)
  {
    const int r = tid >> 3, seg = tid & 7;
    const float4* rp =
        reinterpret_cast<const float4*>(residual + (long)(rowBlk + r) * HID +
                                        seg * 64);
    float s = 0.f;
    #pragma unroll 4
    for (int i = 0; i < 16; ++i) {
      float4 v = rp[i];
      s += v.x * v.x + v.y * v.y + v.z * v.z + v.w * v.w;
    }
    s += __shfl_xor(s, 1, 32);
    s += __shfl_xor(s, 2, 32);
    s += __shfl_xor(s, 4, 32);
    if (seg == 0) s_rn[r] = s;
  }
  __syncthreads();

  const int waveM = wave >> 2, waveN = wave & 3;  // 2x4 tiles -> 32x128 cols
  const int rowT = rowBlk + waveM * 16;
  const __bf16* arh = res_hi + (long)(rowT + l15) * HID;
  const __bf16* arl = res_lo + (long)(rowT + l15) * HID;
  const __bf16* cbh = cb_hi + (long)level * KCB * HID;
  const __bf16* cbl = cb_lo + (long)level * KCB * HID;
  const float* cn = cnorm + level * KCB;

  for (int ct = 0; ct < KCB / 128; ++ct) {
    const int cw = ct * 128 + waveN * 16 + l15;
    const __bf16* brh = cbh + (long)cw * HID;
    const __bf16* brl = cbl + (long)cw * HID;
    v8f acc = {};
    #pragma unroll 2
    for (int k0 = 0; k0 < HID; k0 += 32) {
      v16bf ah = load_a16(arh, k0, half);
      v16bf al = load_a16(arl, k0, half);
      v16bf bh = load_b16(brh, k0, half);
      v16bf bl = load_b16(brl, k0, half);
      acc = __builtin_amdgcn_wmma_f32_16x16x32_bf16(
          false, ah, false, bh, (short)0, acc, false, false);
      acc = __builtin_amdgcn_wmma_f32_16x16x32_bf16(
          false, ah, false, bl, (short)0, acc, false, false);
      acc = __builtin_amdgcn_wmma_f32_16x16x32_bf16(
          false, al, false, bh, (short)0, acc, false, false);
    }
    const float cnv = cn[cw];
    #pragma unroll
    for (int j = 0; j < 8; ++j) {
      const int rloc = waveM * 16 + j + 8 * half;  // row within block's 32
      const float d2 = s_rn[rloc] + cnv - 2.0f * acc[j];
      const float dist = sqrtf(fmaxf(d2, 0.f));
      logits[(long)(rowBlk + rloc) * (LQ * KCB) + level * KCB + cw] = -dist;
      // argmin key: dist>=0 so f32 bits are order-preserving; idx breaks ties
      unsigned khi = __float_as_uint(dist);
      unsigned klo = (unsigned)cw;
      #pragma unroll
      for (int off = 1; off < 16; off <<= 1) {  // stays within 16-lane half
        unsigned ohi = __shfl_xor(khi, off, 32);
        unsigned olo = __shfl_xor(klo, off, 32);
        if (ohi < khi || (ohi == khi && olo < klo)) { khi = ohi; klo = olo; }
      }
      if (l15 == 0)
        atomicMin(&s_key[rloc], ((unsigned long long)khi << 32) | klo);
    }
  }
  __syncthreads();

  // gather winning codewords (exact f32) + residual/qsum update
  const float* cbL = codebooks + (long)level * KCB * HID;
  for (int e = tid; e < 32 * HID; e += 256) {
    const int r = e >> 9, d = e & (HID - 1);
    const unsigned idx = (unsigned)(s_key[r] & 0xffffffffull);
    const float q = cbL[(long)idx * HID + d];
    const long g = (long)(rowBlk + r) * HID + d;
    const float rn = residual[g] - q;
    residual[g] = rn;
    __bf16 h = (__bf16)rn;
    res_hi[g] = h;
    res_lo[g] = (__bf16)(rn - (float)h);
    if (level == 0) qsum[g] = q;
    else            qsum[g] += q;
  }
}

extern "C" void kernel_launch(void* const* d_in, const int* in_sizes, int n_in,
                              void* d_out, int out_size, void* d_ws,
                              size_t ws_size, hipStream_t stream) {
  const float* features  = (const float*)d_in[0];  // [B][HID]
  const float* W_proj    = (const float*)d_in[1];  // [HID][HID]
  const float* b_proj    = (const float*)d_in[2];  // [HID]
  const float* codebooks = (const float*)d_in[3];  // [L][K][HID]

  float* logits = (float*)d_out;                              // [B][L][K]
  float* qsum   = (float*)d_out + (long)BATCH * LQ * KCB;     // [B][HID]

  // workspace layout (all regions 32B-aligned by construction)
  char* p = (char*)d_ws;
  float*  residual = (float*)p;  p += (long)BATCH * HID * 4;
  __bf16* f_hi     = (__bf16*)p; p += (long)BATCH * HID * 2;
  __bf16* f_lo     = (__bf16*)p; p += (long)BATCH * HID * 2;
  __bf16* res_hi   = (__bf16*)p; p += (long)BATCH * HID * 2;
  __bf16* res_lo   = (__bf16*)p; p += (long)BATCH * HID * 2;
  __bf16* w_hi     = (__bf16*)p; p += (long)HID * HID * 2;
  __bf16* w_lo     = (__bf16*)p; p += (long)HID * HID * 2;
  __bf16* cb_hi    = (__bf16*)p; p += (long)LQ * KCB * HID * 2;
  __bf16* cb_lo    = (__bf16*)p; p += (long)LQ * KCB * HID * 2;
  float*  cnorm    = (float*)p;  p += (long)LQ * KCB * 4;

  convert_hilo_kernel<<<(BATCH * HID) / 256, 256, 0, stream>>>(
      features, f_hi, f_lo, BATCH * HID);
  convert_hilo_kernel<<<(HID * HID) / 256, 256, 0, stream>>>(
      W_proj, w_hi, w_lo, HID * HID);
  convert_hilo_kernel<<<(LQ * KCB * HID) / 256, 256, 0, stream>>>(
      codebooks, cb_hi, cb_lo, LQ * KCB * HID);
  cnorm_kernel<<<(LQ * KCB) / 8, 256, 0, stream>>>(codebooks, cnorm);

  proj_kernel<<<dim3(BATCH / 32, HID / 128), 256, 0, stream>>>(
      f_hi, f_lo, w_hi, w_lo, b_proj, residual, res_hi, res_lo);

  for (int l = 0; l < LQ; ++l) {
    rvq_level_kernel<<<BATCH / 32, 256, 0, stream>>>(
        codebooks, cb_hi, cb_lo, cnorm, residual, res_hi, res_lo, logits,
        qsum, l);
  }
}
